// VectorQuantizerEMA_29600914604835
// MI455X (gfx1250) — matrix operand order repaired and checked by reading, not compiled, not run on hardware
//
#include <hip/hip_runtime.h>
#include <hip/hip_bf16.h>

typedef __attribute__((ext_vector_type(16))) _Float16 v16h;
typedef __attribute__((ext_vector_type(8)))  float    v8f;

union Frag {
  v16h     v;
  uint32_t u[8];
  uint4    q[2];
};

#define NUM_EMB 1024
#define EMB_DIM 256
#define NROWS   65536
#define TOTAL   16777216   // N * C
#define DHW     8192
#define DECAYF  0.6f
#define EPSF    1e-5f

// ---- workspace byte offsets ----
#define WS_IDX 0                        // int32 [65536]
#define WS_BH  (WS_IDX + 65536*4)       // u32   [131072]  f16-hi packed embedding
#define WS_BL  (WS_BH + 131072*4)       // u32   [131072]  f16-lo packed embedding
#define WS_EN2 (WS_BL + 131072*4)       // f32   [1024]    ||e||^2
#define WS_CNT (WS_EN2 + 1024*4)        // f32   [1024]    counts
#define WS_ES  (WS_CNT + 1024*4)        // f32   [262144]  embed_sum
#define WS_LP  (WS_ES + 262144*4)       // f32   [8192]    loss partials

// ---- d_out float offsets (tuple order) ----
#define OUT_Q    ((size_t)0)
#define OUT_LOSS ((size_t)16777216)
#define OUT_IDX  ((size_t)16777217)
#define OUT_PERP ((size_t)16842753)
#define OUT_USED ((size_t)16842754)
#define OUT_NEMB ((size_t)16842755)
#define OUT_NCS  ((size_t)17104899)
#define OUT_NAVG ((size_t)17105923)

// ---------------------------------------------------------------------------
// Prep: ||e||^2 (fp32) and f16 hi/lo split of embedding, pre-packed into the
// exact B-fragment layout (k = (j<4 ? 2j : 8+2j) + 8*half, lane-major: 8 dwords contiguous
// per lane -> ds_read_b128 in the GEMM).
// ---------------------------------------------------------------------------
__global__ void k_prep(const float* __restrict__ emb, uint32_t* __restrict__ bh,
                       uint32_t* __restrict__ bl, float* __restrict__ en2) {
  int m = blockIdx.x;          // code index
  int p = threadIdx.x;         // 0..127, channel pair
  int c0 = p * 2;
  float v0 = emb[m * EMB_DIM + c0];
  float v1 = emb[m * EMB_DIM + c0 + 1];
  _Float16 h0 = (_Float16)v0, h1 = (_Float16)v1;
  _Float16 l0 = (_Float16)(v0 - (float)h0), l1 = (_Float16)(v1 - (float)h1);
  union { _Float16 h[2]; uint32_t u; } ph, pl;
  ph.h[0] = h0; ph.h[1] = h1;
  pl.h[0] = l0; pl.h[1] = l1;
  int t  = m >> 4, r = m & 15;       // tile, col-in-tile
  int kt = c0 >> 5;                  // k-step
  int kk = c0 & 31;
  int hf = (kk >> 3) & 1;            // B layout: lanes16-31 hold K=8..15,24..31
  int j  = ((kk & 7) + ((kk >> 4) << 3)) >> 1;  // VGPR index
  int di = t * 2048 + kt * 256 + hf * 128 + r * 8 + j;
  bh[di] = ph.u;
  bl[di] = pl.u;

  __shared__ float red[128];
  red[p] = v0 * v0 + v1 * v1;
  __syncthreads();
  for (int s = 64; s > 0; s >>= 1) {
    if (p < s) red[p] += red[p + s];
    __syncthreads();
  }
  if (p == 0) en2[m] = red[0];
}

// ---------------------------------------------------------------------------
// Main GEMM + argmin. 8 waves / block, 16 rows / wave, 128 rows / block.
// f16x3 split: dot = Ahi*Bhi + Ahi*Blo + Alo*Bhi (fp32 accumulate).
// ---------------------------------------------------------------------------
__global__ void __launch_bounds__(256) k_argmin(
    const float* __restrict__ z, const uint32_t* __restrict__ bh,
    const uint32_t* __restrict__ bl, const float* __restrict__ en2,
    int* __restrict__ idx_out, float* __restrict__ idxf_out) {
  __shared__ __align__(16) uint32_t sB[16384];  // 64KB: [0..8191]=hi, [8192..]=lo

  int tid  = threadIdx.x;
  int w    = tid >> 5;
  int lane = tid & 31;
  int r    = lane & 15;
  int hf   = lane >> 4;
  int base = blockIdx.x * 128 + w * 16;
  int n    = base + r;                 // this lane's row (both halves share it)
  int b    = n >> 13;
  int s    = n & 8191;
  const float* zb = z + (size_t)b * EMB_DIM * DHW + s;

  // Load A (16x32 f16 layout): k = (j<4 ? 2j : 8+2j) + 8*half + 32*kt
  Frag Ah[8], Al[8];
#pragma unroll
  for (int kt = 0; kt < 8; ++kt) {
#pragma unroll
    for (int j = 0; j < 8; ++j) {
      int kk = ((j < 4) ? 2 * j : 8 + 2 * j) + 8 * hf;
      int c0 = kt * 32 + kk;
      float v0 = zb[(size_t)c0 * DHW];
      float v1 = zb[(size_t)(c0 + 1) * DHW];
      _Float16 h0 = (_Float16)v0, h1 = (_Float16)v1;
      _Float16 l0 = (_Float16)(v0 - (float)h0), l1 = (_Float16)(v1 - (float)h1);
      union { _Float16 h[2]; uint32_t u; } ph, pl;
      ph.h[0] = h0; ph.h[1] = h1;
      pl.h[0] = l0; pl.h[1] = l1;
      Ah[kt].u[j] = ph.u;
      Al[kt].u[j] = pl.u;
    }
  }

  float bd[8];
  int   bi[8];
#pragma unroll
  for (int v = 0; v < 8; ++v) { bd[v] = 3.4e38f; bi[v] = 0; }

  for (int ch = 0; ch < 16; ++ch) {        // 16 chunks x 4 tiles = 1024 codes
    __syncthreads();
    {
      const uint4* gh = (const uint4*)(bh + ch * 8192);
      const uint4* gl = (const uint4*)(bl + ch * 8192);
      uint4* dh = (uint4*)sB;
      uint4* dl = (uint4*)(sB + 8192);
      for (int i = tid; i < 2048; i += 256) { dh[i] = gh[i]; dl[i] = gl[i]; }
    }
    __syncthreads();

    for (int tl = 0; tl < 4; ++tl) {
      v8f acc = {0.f, 0.f, 0.f, 0.f, 0.f, 0.f, 0.f, 0.f};
#pragma unroll
      for (int kt = 0; kt < 8; ++kt) {
        Frag Bh, Bl;
        int off = tl * 2048 + kt * 256 + lane * 8;
        Bh.q[0] = *(const uint4*)&sB[off];
        Bh.q[1] = *(const uint4*)&sB[off + 4];
        Bl.q[0] = *(const uint4*)&sB[8192 + off];
        Bl.q[1] = *(const uint4*)&sB[8192 + off + 4];
        acc = __builtin_amdgcn_wmma_f32_16x16x32_f16(false, Al[kt].v, false, Bh.v,
                                                     (short)0, acc, false, false);
        acc = __builtin_amdgcn_wmma_f32_16x16x32_f16(false, Ah[kt].v, false, Bl.v,
                                                     (short)0, acc, false, false);
        acc = __builtin_amdgcn_wmma_f32_16x16x32_f16(false, Ah[kt].v, false, Bh.v,
                                                     (short)0, acc, false, false);
      }
      int m0   = (ch * 4 + tl) * 16;
      float en = en2[m0 + r];
      int cand = m0 + r;
#pragma unroll
      for (int v = 0; v < 8; ++v) {
        float d = fmaf(-2.f, acc[v], en);   // ||e||^2 - 2 x.e  (||x||^2 const)
        if (d < bd[v]) { bd[v] = d; bi[v] = cand; }  // strict '<' keeps lowest m
      }
    }
  }

  // argmin across the 16 lanes of each half (tie -> lower index, like argmin)
#pragma unroll
  for (int v = 0; v < 8; ++v) {
    float d = bd[v];
    int   i = bi[v];
#pragma unroll
    for (int m = 1; m < 16; m <<= 1) {
      float od = __shfl_xor(d, m);
      int   oi = __shfl_xor(i, m);
      if (od < d || (od == d && oi < i)) { d = od; i = oi; }
    }
    bd[v] = d; bi[v] = i;
  }
  if (r == 0) {
#pragma unroll
    for (int v = 0; v < 8; ++v) {
      int row = base + v + 8 * hf;    // C/D layout: M = v + 8*half
      idx_out[row]  = bi[v];
      idxf_out[row] = (float)bi[v];
    }
  }
}

// ---------------------------------------------------------------------------
// Gather quantized (in z's channel-major layout), loss partials, counts,
// embed_sum (segment sum of flat rows).
// ---------------------------------------------------------------------------
__global__ void __launch_bounds__(256) k_quant(
    const float* __restrict__ z, const float* __restrict__ emb,
    const int* __restrict__ idx, float* __restrict__ outq,
    float* __restrict__ embed_sum, float* __restrict__ counts,
    float* __restrict__ loss_part) {
  int tid = threadIdx.x;
  size_t i0 = (size_t)blockIdx.x * 2048 + tid;
  float ls = 0.f;
  for (int it = 0; it < 8; ++it) {
    size_t i = i0 + (size_t)it * 256;
    int s = (int)(i & 8191);
    int c = (int)((i >> 13) & 255);
    int b = (int)(i >> 21);
    int n = (b << 13) | s;
    int m = idx[n];
    float zv = z[i];
    float q  = emb[m * EMB_DIM + c];
    outq[i] = q;                       // quantized_st == quantized numerically
    float df = q - zv;
    ls += df * df;
    atomicAdd(&embed_sum[m * EMB_DIM + c], zv);
    if (c == 0) atomicAdd(&counts[m], 1.0f);   // exact in fp32
  }
  __shared__ float red[256];
  red[tid] = ls;
  __syncthreads();
  for (int st = 128; st > 0; st >>= 1) {
    if (tid < st) red[tid] += red[tid + st];
    __syncthreads();
  }
  if (tid == 0) loss_part[blockIdx.x] = red[0];
}

// ---------------------------------------------------------------------------
// EMA row updates: new_embedding_avg, new_embedding, new_cluster_size
// ---------------------------------------------------------------------------
__global__ void k_rows(const float* __restrict__ eavg, const float* __restrict__ cs,
                       const float* __restrict__ counts, const float* __restrict__ es,
                       float* __restrict__ out_nemb, float* __restrict__ out_ncs,
                       float* __restrict__ out_navg) {
  int m = blockIdx.x, c = threadIdx.x;
  float ncs  = cs[m] * DECAYF + (1.f - DECAYF) * counts[m];
  float navg = eavg[m * EMB_DIM + c] * DECAYF + (1.f - DECAYF) * es[m * EMB_DIM + c];
  out_navg[m * EMB_DIM + c] = navg;
  out_nemb[m * EMB_DIM + c] = navg / (ncs + EPSF);
  if (c == 0) out_ncs[m] = ncs;
}

// ---------------------------------------------------------------------------
// Scalars: loss, perplexity, used_codes (deterministic fixed-order trees)
// ---------------------------------------------------------------------------
__global__ void k_scalars(const float* __restrict__ counts, const float* __restrict__ cs,
                          const float* __restrict__ loss_part, float* __restrict__ out_loss,
                          float* __restrict__ out_perp, float* __restrict__ out_used) {
  int t = threadIdx.x;  // 1024
  float a = counts[t] * (1.0f / (float)NROWS);
  float p = a * logf(a + 1e-10f);
  float u = (cs[t] > 1e-5f) ? 1.f : 0.f;
  float l = 0.f;
  for (int i = t; i < 8192; i += 1024) l += loss_part[i];
  __shared__ float rp[1024], ru[1024], rl[1024];
  rp[t] = p; ru[t] = u; rl[t] = l;
  __syncthreads();
  for (int s = 512; s > 0; s >>= 1) {
    if (t < s) { rp[t] += rp[t + s]; ru[t] += ru[t + s]; rl[t] += rl[t + s]; }
    __syncthreads();
  }
  if (t == 0) {
    *out_loss = rl[0] * (1.0f / (float)TOTAL);
    *out_perp = expf(-rp[0]);
    *out_used = ru[0] * (1.0f / (float)NUM_EMB);
  }
}

extern "C" void kernel_launch(void* const* d_in, const int* in_sizes, int n_in,
                              void* d_out, int out_size, void* d_ws, size_t ws_size,
                              hipStream_t stream) {
  (void)in_sizes; (void)n_in; (void)out_size; (void)ws_size;
  const float* z    = (const float*)d_in[0];
  const float* emb  = (const float*)d_in[1];
  const float* cs   = (const float*)d_in[2];
  const float* eavg = (const float*)d_in[3];
  float* out = (float*)d_out;
  char*  ws  = (char*)d_ws;

  int*      idx = (int*)(ws + WS_IDX);
  uint32_t* bhp = (uint32_t*)(ws + WS_BH);
  uint32_t* blp = (uint32_t*)(ws + WS_BL);
  float*    en2 = (float*)(ws + WS_EN2);
  float*    cnt = (float*)(ws + WS_CNT);
  float*    es  = (float*)(ws + WS_ES);
  float*    lp  = (float*)(ws + WS_LP);

  hipMemsetAsync(cnt, 0, 1024 * 4, stream);
  hipMemsetAsync(es, 0, 262144 * 4, stream);

  k_prep<<<1024, 128, 0, stream>>>(emb, bhp, blp, en2);
  k_argmin<<<512, 256, 0, stream>>>(z, bhp, blp, en2, idx, out + OUT_IDX);
  k_quant<<<8192, 256, 0, stream>>>(z, emb, idx, out + OUT_Q, es, cnt, lp);
  k_rows<<<1024, 256, 0, stream>>>(eavg, cs, cnt, es, out + OUT_NEMB,
                                   out + OUT_NCS, out + OUT_NAVG);
  k_scalars<<<1, 1024, 0, stream>>>(cnt, cs, lp, out + OUT_LOSS,
                                    out + OUT_PERP, out + OUT_USED);
}
